// SCReGAT_26749056320130
// MI455X (gfx1250) — matrix operand order repaired and checked by reading, not compiled, not run on hardware
//
#include <hip/hip_runtime.h>
#include <math.h>

typedef __attribute__((ext_vector_type(16))) _Float16 v16h;
typedef __attribute__((ext_vector_type(8)))  _Float16 v8h;
typedef __attribute__((ext_vector_type(8)))  float    v8f;

static __device__ __forceinline__ float leakyf(float x, float s) {
    return fmaxf(x, 0.f) + s * fminf(x, 0.f);
}

static __device__ __forceinline__ float atomicMaxFloat(float* addr, float val) {
    unsigned int* ua = (unsigned int*)addr;
    unsigned int old = *ua;
    while (true) {
        float f = __uint_as_float(old);
        if (f >= val) break;
        unsigned int assumed = old;
        old = atomicCAS(ua, assumed, __float_as_uint(val));
        if (old == assumed) break;
    }
    return __uint_as_float(old);
}

// ---------------------------------------------------------------------------
// A-fragment 8-element chunk loader. All K used here are multiples of 8 and
// chunks are 8-aligned, so validity is per-chunk: two b128 loads, no
// per-element branches.  IN_MODE: 0 = raw, 1 = scale*x+shift then leaky,
// 2 = x+shift.
// ---------------------------------------------------------------------------
template<int IN_MODE>
static __device__ __forceinline__ void load_chunk8(const float* __restrict__ Arow,
                                                   int k, int K, bool mOK,
                                                   const float* __restrict__ sc,
                                                   const float* __restrict__ sh,
                                                   float slope,
                                                   float* __restrict__ dst) {
    if (mOK && k < K) {
        float4 a = *(const float4*)(Arow + k);
        float4 b = *(const float4*)(Arow + k + 4);
        float v[8] = {a.x, a.y, a.z, a.w, b.x, b.y, b.z, b.w};
        if (IN_MODE == 1) {
            float4 s0 = *(const float4*)(sc + k);
            float4 s1 = *(const float4*)(sc + k + 4);
            float4 t0 = *(const float4*)(sh + k);
            float4 t1 = *(const float4*)(sh + k + 4);
            float ss[8] = {s0.x, s0.y, s0.z, s0.w, s1.x, s1.y, s1.z, s1.w};
            float tt[8] = {t0.x, t0.y, t0.z, t0.w, t1.x, t1.y, t1.z, t1.w};
            #pragma unroll
            for (int i = 0; i < 8; ++i) {
                float x = ss[i] * v[i] + tt[i];
                v[i] = leakyf(x, slope);
            }
        } else if (IN_MODE == 2) {
            float4 t0 = *(const float4*)(sh + k);
            float4 t1 = *(const float4*)(sh + k + 4);
            float tt[8] = {t0.x, t0.y, t0.z, t0.w, t1.x, t1.y, t1.z, t1.w};
            #pragma unroll
            for (int i = 0; i < 8; ++i) v[i] += tt[i];
        }
        #pragma unroll
        for (int i = 0; i < 8; ++i) dst[i] = v[i];
    } else {
        #pragma unroll
        for (int i = 0; i < 8; ++i) dst[i] = 0.f;
    }
}

// ---------------------------------------------------------------------------
// Generic WMMA GEMM: Out[M,ncValid] = act_out( act_in(A) @ W + bias )
//   A: [M,K] row-major f32 (streamed as b128 loads, f32->f16 on the fly).
//   W: [K,ncValid] row-major f32, staged once into LDS as f16 pre-swizzled
//   into B-fragment layout: [kChunk][colTile][lane][16 halves], each lane's
//   fragment padded to 24 halves (48B stride -> conflict-free ds_load_b128).
//   One wave per 16x16 output tile; v_wmma_f32_16x16x32_f16 over K chunks.
// ---------------------------------------------------------------------------
#define FRAG_STRIDE 24   // halves per lane fragment (16 used + 8 pad)

template<int IN_MODE, bool HAS_BIAS, bool OUT_LEAKY>
__global__ void gemm_wmma_kernel(const float* __restrict__ A, int M, int K,
                                 const float* __restrict__ inScale,
                                 const float* __restrict__ inShift,
                                 float slopeIn,
                                 const float* __restrict__ W, int ncValid,
                                 const float* __restrict__ bias,
                                 float slopeOut,
                                 float* __restrict__ Out, int ldOut) {
    extern __shared__ _Float16 Wh[];
    const int nTiles  = blockDim.x >> 5;
    const int K32     = (K + 31) & ~31;
    const int nChunks = K32 >> 5;

    // Cooperative stage: f32 weights -> f16 B-fragments in LDS.
    // Element (chunk ch, tile t, lane l, j) holds W[ch*32 + (l<16?0:8) + (j<8?j:j+8)][t*16 + (l&15)].
    const int tflat = threadIdx.y * blockDim.x + threadIdx.x;
    const int nThr  = blockDim.x * blockDim.y;
    const int total = nChunks * nTiles * 32 * 16;
    for (int idx = tflat; idx < total; idx += nThr) {
        int j    = idx & 15;
        int l    = (idx >> 4) & 31;
        int ft   = (idx >> 9);
        int t    = ft % nTiles;
        int ch   = ft / nTiles;
        int k    = ch * 32 + ((l < 16) ? 0 : 8) + ((j < 8) ? j : j + 8);
        int col  = t * 16 + (l & 15);
        float w  = (k < K && col < ncValid) ? W[k * ncValid + col] : 0.f;
        Wh[(size_t)(ft * 32 + l) * FRAG_STRIDE + j] = (_Float16)w;
    }
    __syncthreads();

    const int wave = threadIdx.x >> 5;
    const int lane = threadIdx.x & 31;
    const int row0 = (blockIdx.x * blockDim.y + threadIdx.y) * 16;
    if (row0 >= M) return;                 // uniform per wave

    v8f acc = {};
    const int  m   = row0 + (lane & 15);
    const bool mOK = (m < M);
    const float* Arow = A + (long long)m * K;
    const int col = (wave << 4) + (lane & 15);
    const _Float16* fragBase = Wh + (size_t)(wave * 32 + lane) * FRAG_STRIDE;
    const size_t chunkStride = (size_t)nTiles * 32 * FRAG_STRIDE;

    for (int ch = 0; ch < nChunks; ++ch) {
        const int k0 = ch * 32;
        // ---- A fragment (16-bit A 16x32 layout) ----
        float xa[16];
        const int kb0 = k0 + ((lane < 16) ? 0 : 8);
        load_chunk8<IN_MODE>(Arow, kb0,      K, mOK, inScale, inShift, slopeIn, xa);
        load_chunk8<IN_MODE>(Arow, kb0 + 16, K, mOK, inScale, inShift, slopeIn, xa + 8);
        v16h a;
        #pragma unroll
        for (int j = 0; j < 16; ++j) a[j] = (_Float16)xa[j];

        // ---- B fragment: two ds_load_b128 of pre-swizzled f16 ----
        const _Float16* fb = fragBase + ch * chunkStride;
        v8h b0 = *(const v8h*)fb;
        v8h b1 = *(const v8h*)(fb + 8);
        v16h b;
        #pragma unroll
        for (int j = 0; j < 8; ++j) { b[j] = b0[j]; b[8 + j] = b1[j]; }

        acc = __builtin_amdgcn_wmma_f32_16x16x32_f16(
                  false, a, false, b, (short)0, acc, false, false);
    }

    if (col < ncValid) {
        float bv = HAS_BIAS ? bias[col] : 0.f;
        int rbase = row0 + ((lane < 16) ? 0 : 8);
        #pragma unroll
        for (int i = 0; i < 8; ++i) {
            int r = rbase + i;
            if (r < M) {
                float v = acc[i] + bv;
                if (OUT_LEAKY) v = leakyf(v, slopeOut);
                Out[(long long)r * ldOut + col] = v;
            }
        }
    }
}

// ---------------------------------------------------------------------------
// Small utility kernels
// ---------------------------------------------------------------------------
__global__ void fill_kernel(float* p, float v, long long n) {
    long long t = (long long)blockIdx.x * blockDim.x + threadIdx.x;
    if (t < n) p[t] = v;
}

__global__ void add_kernel(float* a, const float* b, long long n) {
    long long t = (long long)blockIdx.x * blockDim.x + threadIdx.x;
    if (t < n) a[t] += b[t];
}

// pre1[n,128] = raw_x[n] * nW1[j] + nb1[j]
__global__ void node_l1_kernel(const float* __restrict__ raw_x,
                               const float* __restrict__ W,
                               const float* __restrict__ b,
                               int N, float* __restrict__ pre1) {
    long long t = (long long)blockIdx.x * blockDim.x + threadIdx.x;
    if (t >= (long long)N * 128) return;
    int n = (int)(t >> 7), j = (int)(t & 127);
    pre1[t] = raw_x[n] * W[j] + b[j];
}

// per-channel sum / sumsq of x[M,C]  (C divides blockDim.x)
__global__ void bn_stats_kernel(const float* __restrict__ x, long long M, int C,
                                float* __restrict__ sums) {
    int tid = blockIdx.x * blockDim.x + threadIdx.x;
    int j = tid % C;
    long long r0 = tid / C;
    long long stride = (long long)(gridDim.x * blockDim.x) / C;
    float s = 0.f, ss = 0.f;
    for (long long r = r0; r < M; r += stride) {
        float v = x[r * C + j];
        s += v; ss += v * v;
    }
    atomicAdd(&sums[j], s);
    atomicAdd(&sums[C + j], ss);
}

__global__ void bn_finalize_kernel(const float* __restrict__ sums,
                                   const float* __restrict__ g,
                                   const float* __restrict__ b,
                                   int C, float invM,
                                   float* __restrict__ scale,
                                   float* __restrict__ shift) {
    int j = blockIdx.x * blockDim.x + threadIdx.x;
    if (j >= C) return;
    float m  = sums[j] * invM;
    float v  = sums[C + j] * invM - m * m;
    float sc = g[j] * rsqrtf(v + 1e-5f);
    scale[j] = sc;
    shift[j] = b[j] - m * sc;
}

// ---------------------------------------------------------------------------
// Edge MLP (3 -> 32 -> 16 -> 8), BN stats fused, layer1 recomputed on the fly
// ---------------------------------------------------------------------------
__global__ void edge_l1_stats_kernel(const float* __restrict__ raw_x,
                                     const int* __restrict__ ei, int E,
                                     const float* __restrict__ W1,
                                     const float* __restrict__ b1,
                                     float* __restrict__ sums /*64*/) {
    int tid = blockIdx.x * blockDim.x + threadIdx.x;
    int j = tid & 31;
    long long r0 = tid >> 5;
    long long stride = ((long long)gridDim.x * blockDim.x) >> 5;
    float s = 0.f, ss = 0.f;
    for (long long e = r0; e < E; e += stride) {
        float xr = raw_x[ei[e]];
        float xc = raw_x[ei[E + e]];
        float h = (xr * xc) * W1[j] + xr * W1[32 + j] + xc * W1[64 + j] + b1[j];
        s += h; ss += h * h;
    }
    atomicAdd(&sums[j], s);
    atomicAdd(&sums[32 + j], ss);
}

__global__ void edge_l2_kernel(const float* __restrict__ raw_x,
                               const int* __restrict__ ei, int E,
                               const float* __restrict__ W1, const float* __restrict__ b1,
                               const float* __restrict__ s1, const float* __restrict__ h1,
                               const float* __restrict__ W2, const float* __restrict__ b2,
                               float* __restrict__ h2out, float* __restrict__ sums /*32*/) {
    float ls[16], lss[16];
    #pragma unroll
    for (int c = 0; c < 16; ++c) { ls[c] = 0.f; lss[c] = 0.f; }
    for (long long e = (long long)blockIdx.x * blockDim.x + threadIdx.x; e < E;
         e += (long long)gridDim.x * blockDim.x) {
        float xr = raw_x[ei[e]];
        float xc = raw_x[ei[E + e]];
        float f0 = xr * xc;
        float a1[32];
        #pragma unroll
        for (int j = 0; j < 32; ++j) {
            float h = f0 * W1[j] + xr * W1[32 + j] + xc * W1[64 + j] + b1[j];
            h = s1[j] * h + h1[j];
            a1[j] = leakyf(h, 0.01f);
        }
        #pragma unroll
        for (int c = 0; c < 16; ++c) {
            float h = b2[c];
            #pragma unroll
            for (int j = 0; j < 32; ++j) h += a1[j] * W2[j * 16 + c];
            h2out[e * 16 + c] = h;
            ls[c] += h; lss[c] += h * h;
        }
    }
    #pragma unroll
    for (int c = 0; c < 16; ++c) {
        atomicAdd(&sums[c], ls[c]);
        atomicAdd(&sums[16 + c], lss[c]);
    }
}

__global__ void edge_l3_kernel(const float* __restrict__ h2, long long E,
                               const float* __restrict__ s2, const float* __restrict__ sh2,
                               const float* __restrict__ W3, const float* __restrict__ b3,
                               float* __restrict__ h3out, float* __restrict__ sums /*16*/) {
    float ls[8], lss[8];
    #pragma unroll
    for (int c = 0; c < 8; ++c) { ls[c] = 0.f; lss[c] = 0.f; }
    for (long long e = (long long)blockIdx.x * blockDim.x + threadIdx.x; e < E;
         e += (long long)gridDim.x * blockDim.x) {
        float a2[16];
        #pragma unroll
        for (int j = 0; j < 16; ++j) {
            float h = s2[j] * h2[e * 16 + j] + sh2[j];
            a2[j] = leakyf(h, 0.01f);
        }
        #pragma unroll
        for (int c = 0; c < 8; ++c) {
            float h = b3[c];
            #pragma unroll
            for (int j = 0; j < 16; ++j) h += a2[j] * W3[j * 8 + c];
            h3out[e * 8 + c] = h;
            ls[c] += h; lss[c] += h * h;
        }
    }
    #pragma unroll
    for (int c = 0; c < 8; ++c) {
        atomicAdd(&sums[c], ls[c]);
        atomicAdd(&sums[8 + c], lss[c]);
    }
}

__global__ void edge_final_kernel(float* __restrict__ h3, long long n,
                                  const float* __restrict__ s3,
                                  const float* __restrict__ sh3) {
    long long t = (long long)blockIdx.x * blockDim.x + threadIdx.x;
    if (t >= n) return;
    int c = (int)(t & 7);
    float h = s3[c] * h3[t] + sh3[c];
    h3[t] = tanhf(leakyf(h, 0.01f));
}

// ---------------------------------------------------------------------------
// Composite weight builders:  comp[K,288] = [ W(256) | W.as(16) | W.ad(16) ]
// ---------------------------------------------------------------------------
__global__ void copy_cols_kernel(const float* __restrict__ src, int K, int nc,
                                 float* __restrict__ dst, int ldDst, int colOff) {
    int t = blockIdx.x * blockDim.x + threadIdx.x;
    if (t >= K * nc) return;
    int k = t / nc, c = t - k * nc;
    dst[k * ldDst + colOff + c] = src[t];
}

// dst[k, colOff+h] = sum_c W[k, h*16+c] * ah[h*16+c]
__global__ void proj_kernel(const float* __restrict__ W, int K,
                            const float* __restrict__ ah,
                            float* __restrict__ dst, int ldDst, int colOff) {
    int t = blockIdx.x * blockDim.x + threadIdx.x;
    if (t >= K * 16) return;
    int k = t >> 4, h = t & 15;
    float s = 0.f;
    #pragma unroll
    for (int c = 0; c < 16; ++c) s += W[k * 256 + h * 16 + c] * ah[h * 16 + c];
    dst[k * ldDst + colOff + h] = s;
}

// ---------------------------------------------------------------------------
// GAT segment softmax + aggregation
// xw: [N,288] = [ xW(256) | alpha_src(16) | alpha_dst(16) ]
// ---------------------------------------------------------------------------
__global__ void gat_alpha_pre_kernel(const float* __restrict__ xw,
                                     const int* __restrict__ ei, int E,
                                     const float* __restrict__ aEadd,
                                     float* __restrict__ aW,
                                     float* __restrict__ amax) {
    int t = blockIdx.x * blockDim.x + threadIdx.x;
    if (t >= E * 16) return;
    int e = t >> 4, h = t & 15;
    int r = ei[e], c = ei[E + e];
    float v = xw[(long long)r * 288 + 256 + h] + xw[(long long)c * 288 + 272 + h];
    if (aEadd) v += aEadd[t];
    v = leakyf(v, 0.2f);
    aW[t] = v;
    atomicMaxFloat(&amax[c * 16 + h], v);
}

__global__ void gat_exp_kernel(float* __restrict__ aW,
                               const int* __restrict__ ei, int E,
                               const float* __restrict__ amax,
                               float* __restrict__ denom) {
    int t = blockIdx.x * blockDim.x + threadIdx.x;
    if (t >= E * 16) return;
    int e = t >> 4, h = t & 15;
    int c = ei[E + e];
    float v = expf(aW[t] - amax[c * 16 + h]);
    aW[t] = v;
    atomicAdd(&denom[c * 16 + h], v);
}

__global__ void gat_norm_kernel(float* __restrict__ aW,
                                const int* __restrict__ ei, int E,
                                const float* __restrict__ denom,
                                float* __restrict__ alphaOut) {
    int t = blockIdx.x * blockDim.x + threadIdx.x;
    if (t >= E * 16) return;
    int e = t >> 4, h = t & 15;
    int c = ei[E + e];
    float a = aW[t] / (denom[c * 16 + h] + 1e-16f);
    aW[t] = a;
    if (alphaOut) alphaOut[t] = a;
}

// one wave per edge; lane handles 8 consecutive channels (within one head)
__global__ void gat_aggr_kernel(const float* __restrict__ xw,
                                const float* __restrict__ alpha,
                                const int* __restrict__ ei, int E,
                                float* __restrict__ outacc) {
    int wave = (blockIdx.x * blockDim.x + threadIdx.x) >> 5;
    int lane = threadIdx.x & 31;
    if (wave >= E) return;
    int r = ei[wave], c = ei[E + wave];
    int ch = lane * 8;
    float a = alpha[(long long)wave * 16 + (ch >> 4)];
    const float* xr = xw + (long long)r * 288 + ch;
    float* od = outacc + (long long)c * 256 + ch;
    float4 x0 = *(const float4*)xr;
    float4 x1 = *(const float4*)(xr + 4);
    atomicAdd(&od[0], a * x0.x); atomicAdd(&od[1], a * x0.y);
    atomicAdd(&od[2], a * x0.z); atomicAdd(&od[3], a * x0.w);
    atomicAdd(&od[4], a * x1.x); atomicAdd(&od[5], a * x1.y);
    atomicAdd(&od[6], a * x1.z); atomicAdd(&od[7], a * x1.w);
}

// gene_out[n/25] = logsumexp(data[n,:]) - data[n,0] for masked nodes
__global__ void gene_kernel(const float* __restrict__ data,
                            const int* __restrict__ gid, int N,
                            float* __restrict__ out) {
    int n = blockIdx.x * blockDim.x + threadIdx.x;
    if (n >= N) return;
    if (gid[n] != 1) return;
    const float* r = data + (long long)n * 16;
    float m = r[0];
    #pragma unroll
    for (int c = 1; c < 16; ++c) m = fmaxf(m, r[c]);
    float s = 0.f;
    #pragma unroll
    for (int c = 0; c < 16; ++c) s += expf(r[c] - m);
    out[n / 25] = (m + logf(s)) - r[0];
}

// ---------------------------------------------------------------------------
// Host-side launch orchestration
// ---------------------------------------------------------------------------
static inline int cdiv(long long a, long long b) { return (int)((a + b - 1) / b); }

static void launch_gemm(const float* A, int M, int K,
                        const float* inScale, const float* inShift, float slopeIn,
                        const float* W, int ncValid, const float* bias,
                        float slopeOut, float* Out, int ldOut, hipStream_t stream) {
    int nTiles = (ncValid + 15) / 16;
    int bx = nTiles * 32;
    int ry = 1;
    while (bx * ry * 2 <= 256) ry *= 2;
    dim3 blk(bx, ry);
    int rowBlocks = cdiv(M, 16 * ry);
    int K32 = (K + 31) & ~31;
    int nChunks = K32 / 32;
    size_t lds = (size_t)nChunks * nTiles * 32 * FRAG_STRIDE * 2;  // f16 fragments

    int im  = inScale ? 1 : (inShift ? 2 : 0);
    int hb  = (bias != nullptr) ? 1 : 0;
    int ol  = (slopeOut != 1.0f) ? 1 : 0;
    int sel = im * 4 + hb * 2 + ol;

#define GO(IM, HB, OL) gemm_wmma_kernel<IM, HB, OL><<<rowBlocks, blk, lds, stream>>>( \
        A, M, K, inScale, inShift, slopeIn, W, ncValid, bias, slopeOut, Out, ldOut)
    switch (sel) {
        case 0:  GO(0, false, false); break;
        case 1:  GO(0, false, true ); break;
        case 2:  GO(0, true , false); break;
        case 3:  GO(0, true , true ); break;
        case 4:  GO(1, false, false); break;
        case 5:  GO(1, false, true ); break;
        case 6:  GO(1, true , false); break;
        case 7:  GO(1, true , true ); break;
        case 8:  GO(2, false, false); break;
        case 9:  GO(2, false, true ); break;
        case 10: GO(2, true , false); break;
        default: GO(2, true , true ); break;
    }
#undef GO
}

extern "C" void kernel_launch(void* const* d_in, const int* in_sizes, int n_in,
                              void* d_out, int out_size, void* d_ws, size_t ws_size,
                              hipStream_t stream) {
    (void)n_in; (void)out_size; (void)ws_size;

    const float* raw_x = (const float*)d_in[1];
    const int*   ei    = (const int*)d_in[2];
    const int*   etf   = (const int*)d_in[3];
    const int*   gid   = (const int*)d_in[6];
    const int N   = in_sizes[1];
    const int E   = in_sizes[2] / 2;
    const int ETF = in_sizes[3] / 2;

    const float *nW1=(const float*)d_in[7],  *nb1=(const float*)d_in[8],
                *ng1=(const float*)d_in[9],  *nbe1=(const float*)d_in[10],
                *nW2=(const float*)d_in[11], *nb2=(const float*)d_in[12],
                *ng2=(const float*)d_in[13], *nbe2=(const float*)d_in[14],
                *nW3=(const float*)d_in[15], *nb3=(const float*)d_in[16],
                *eW1=(const float*)d_in[17], *eb1=(const float*)d_in[18],
                *eg1=(const float*)d_in[19], *ebe1=(const float*)d_in[20],
                *eW2=(const float*)d_in[21], *eb2=(const float*)d_in[22],
                *eg2=(const float*)d_in[23], *ebe2=(const float*)d_in[24],
                *eW3=(const float*)d_in[25], *eb3=(const float*)d_in[26],
                *eg3=(const float*)d_in[27], *ebe3=(const float*)d_in[28],
                *g1W=(const float*)d_in[29], *g1as=(const float*)d_in[30],
                *g1ad=(const float*)d_in[31], *g1We=(const float*)d_in[32],
                *g1ae=(const float*)d_in[33], *g1b=(const float*)d_in[34],
                *f1W=(const float*)d_in[35], *f1b=(const float*)d_in[36],
                *g2W=(const float*)d_in[37], *g2as=(const float*)d_in[38],
                *g2ad=(const float*)d_in[39], *g2b=(const float*)d_in[40],
                *f2W=(const float*)d_in[41], *f2b=(const float*)d_in[42];

    float* gene_out  = (float*)d_out;
    float* alpha_out = (float*)d_out + 2000;

    // workspace arena (floats); every region size is a multiple of 8 floats
    // so float4 row loads stay 32B-aligned.
    float* ws = (float*)d_ws;
    size_t o = 0;
    auto alloc = [&](size_t n) { float* p = ws + o; o += n; return p; };
    float* pre1   = alloc((size_t)N * 128);
    float* pre2   = alloc((size_t)N * 64);
    float* data0  = alloc((size_t)N * 8);
    float* h2e    = alloc((size_t)E * 16);
    float* hidE   = alloc((size_t)E * 8);
    float* xw     = alloc((size_t)N * 288);
    float* aE     = alloc((size_t)E * 16);
    float* amax   = alloc((size_t)N * 16);
    float* denom  = alloc((size_t)N * 16);
    float* outacc = alloc((size_t)N * 256);
    float* data1  = alloc((size_t)N * 16);
    float* data2  = alloc((size_t)N * 16);
    float* comp1  = alloc(8 * 288);
    float* comp2  = alloc(16 * 288);
    float* Me     = alloc(8 * 16);
    float* sumsAll = alloc(2 * (128 + 64 + 32 + 16 + 8));  // 496
    float* sums1 = sumsAll, *sums2 = sums1 + 256, *sE1 = sums2 + 128,
         * sE2 = sE1 + 64, *sE3 = sE2 + 32;
    float* bnAll = alloc(2 * (128 + 64 + 32 + 16 + 8));    // scale|shift pairs
    float* sc1 = bnAll,      *sh1 = sc1 + 128;
    float* sc2 = sh1 + 128,  *sh2v = sc2 + 64;
    float* se1 = sh2v + 64,  *he1 = se1 + 32;
    float* se2 = he1 + 32,   *he2 = se2 + 16;
    float* se3 = he2 + 16,   *he3 = se3 + 8;

    hipMemsetAsync(sumsAll, 0, 496 * sizeof(float), stream);

    // ---- node MLP ----
    node_l1_kernel<<<cdiv((long long)N * 128, 256), 256, 0, stream>>>(raw_x, nW1, nb1, N, pre1);
    bn_stats_kernel<<<256, 256, 0, stream>>>(pre1, N, 128, sums1);
    bn_finalize_kernel<<<1, 128, 0, stream>>>(sums1, ng1, nbe1, 128, 1.f / N, sc1, sh1);
    launch_gemm(pre1, N, 128, sc1, sh1, 0.01f, nW2, 64, nb2, 1.0f, pre2, 64, stream);
    bn_stats_kernel<<<256, 256, 0, stream>>>(pre2, N, 64, sums2);
    bn_finalize_kernel<<<1, 64, 0, stream>>>(sums2, ng2, nbe2, 64, 1.f / N, sc2, sh2v);
    launch_gemm(pre2, N, 64, sc2, sh2v, 0.01f, nW3, 8, nb3, 1.0f, data0, 8, stream);

    // ---- edge MLP ----
    edge_l1_stats_kernel<<<256, 256, 0, stream>>>(raw_x, ei, E, eW1, eb1, sE1);
    bn_finalize_kernel<<<1, 32, 0, stream>>>(sE1, eg1, ebe1, 32, 1.f / E, se1, he1);
    edge_l2_kernel<<<512, 256, 0, stream>>>(raw_x, ei, E, eW1, eb1, se1, he1, eW2, eb2, h2e, sE2);
    bn_finalize_kernel<<<1, 16, 0, stream>>>(sE2, eg2, ebe2, 16, 1.f / E, se2, he2);
    edge_l3_kernel<<<512, 256, 0, stream>>>(h2e, E, se2, he2, eW3, eb3, hidE, sE3);
    bn_finalize_kernel<<<1, 8, 0, stream>>>(sE3, eg3, ebe3, 8, 1.f / E, se3, he3);
    edge_final_kernel<<<cdiv((long long)E * 8, 256), 256, 0, stream>>>(hidE, (long long)E * 8, se3, he3);

    // ---- composite weights for GAT1 / GAT2 ----
    copy_cols_kernel<<<cdiv(8 * 256, 256), 256, 0, stream>>>(g1W, 8, 256, comp1, 288, 0);
    proj_kernel<<<1, 128, 0, stream>>>(g1W, 8, g1as, comp1, 288, 256);
    proj_kernel<<<1, 128, 0, stream>>>(g1W, 8, g1ad, comp1, 288, 272);
    proj_kernel<<<1, 128, 0, stream>>>(g1We, 8, g1ae, Me, 16, 0);
    copy_cols_kernel<<<cdiv(16 * 256, 256), 256, 0, stream>>>(g2W, 16, 256, comp2, 288, 0);
    proj_kernel<<<1, 256, 0, stream>>>(g2W, 16, g2as, comp2, 288, 256);
    proj_kernel<<<1, 256, 0, stream>>>(g2W, 16, g2ad, comp2, 288, 272);

    // ---- GAT1 ----
    launch_gemm(data0, N, 8, nullptr, nullptr, -1.f, comp1, 288, nullptr, 1.0f, xw, 288, stream);
    launch_gemm(hidE, E, 8, nullptr, nullptr, -1.f, Me, 16, nullptr, 1.0f, aE, 16, stream);
    fill_kernel<<<cdiv((long long)N * 16, 256), 256, 0, stream>>>(amax, -1e30f, (long long)N * 16);
    hipMemsetAsync(denom, 0, (size_t)N * 16 * sizeof(float), stream);
    hipMemsetAsync(outacc, 0, (size_t)N * 256 * sizeof(float), stream);
    gat_alpha_pre_kernel<<<cdiv((long long)E * 16, 256), 256, 0, stream>>>(xw, ei, E, aE, aE, amax);
    gat_exp_kernel<<<cdiv((long long)E * 16, 256), 256, 0, stream>>>(aE, ei, E, amax, denom);
    gat_norm_kernel<<<cdiv((long long)E * 16, 256), 256, 0, stream>>>(aE, ei, E, denom, alpha_out);
    gat_aggr_kernel<<<cdiv((long long)E * 32, 256), 256, 0, stream>>>(xw, aE, ei, E, outacc);
    launch_gemm(outacc, N, 256, nullptr, g1b, -1.f, f1W, 16, f1b, 0.01f, data1, 16, stream);

    // ---- GAT2 (edge_tf, no edge features) ----
    launch_gemm(data1, N, 16, nullptr, nullptr, -1.f, comp2, 288, nullptr, 1.0f, xw, 288, stream);
    fill_kernel<<<cdiv((long long)N * 16, 256), 256, 0, stream>>>(amax, -1e30f, (long long)N * 16);
    hipMemsetAsync(denom, 0, (size_t)N * 16 * sizeof(float), stream);
    hipMemsetAsync(outacc, 0, (size_t)N * 256 * sizeof(float), stream);
    gat_alpha_pre_kernel<<<cdiv((long long)ETF * 16, 256), 256, 0, stream>>>(xw, etf, ETF, nullptr, aE, amax);
    gat_exp_kernel<<<cdiv((long long)ETF * 16, 256), 256, 0, stream>>>(aE, etf, ETF, amax, denom);
    gat_norm_kernel<<<cdiv((long long)ETF * 16, 256), 256, 0, stream>>>(aE, etf, ETF, denom, nullptr);
    gat_aggr_kernel<<<cdiv((long long)ETF * 32, 256), 256, 0, stream>>>(xw, aE, etf, ETF, outacc);
    launch_gemm(outacc, N, 256, nullptr, g2b, -1.f, f2W, 16, f2b, 0.01f, data2, 16, stream);

    // ---- combine + gene output ----
    add_kernel<<<cdiv((long long)N * 16, 256), 256, 0, stream>>>(data1, data2, (long long)N * 16);
    gene_kernel<<<cdiv(N, 256), 256, 0, stream>>>(data1, gid, N, gene_out);
}
